// BioSignalAnomalyDetector_89352499626040
// MI455X (gfx1250) — matrix operand order repaired and checked by reading, not compiled, not run on hardware
//
#include <hip/hip_runtime.h>
#include <hip/hip_bf16.h>
#include <math.h>

// ---------------------------------------------------------------------------
// BioSignalAnomalyDetector: 6-layer transformer encoder + heads, CDNA5 WMMA
// Round 2: vectorized fragment loads (b128), weight pre-transpose, V transpose
// ---------------------------------------------------------------------------

typedef __attribute__((ext_vector_type(16))) __bf16 bf16x16;
typedef __attribute__((ext_vector_type(8)))  __bf16 bf16x8;
typedef __attribute__((ext_vector_type(8)))  float  f32x8;

#define B_   16
#define S_   1000
#define IN_  32
#define D_   512
#define H_   8
#define DK_  64
#define FF_  2048
#define L_   6
#define M_   (B_ * S_)      // 16000 tokens
#define STILES 63           // ceil(1000/16)
#define SPAD  (STILES * 16) // 1008
#define SROW  1024          // padded score row length

__device__ __forceinline__ f32x8 wmma_bf16(bf16x16 a, bf16x16 b, f32x8 c) {
  return __builtin_amdgcn_wmma_f32_16x16x32_bf16(
      false, a, false, b, (short)0, c, false, false);
}

__device__ __forceinline__ bf16x16 cat16(bf16x8 a, bf16x8 b) {
  return __builtin_shufflevector(a, b, 0, 1, 2, 3, 4, 5, 6, 7, 8, 9, 10, 11,
                                 12, 13, 14, 15);
}

// A-fragment (16x32): lane holds row; 16 contiguous K elems at K off hi*16.
// Two aligned 16B chunks.
__device__ __forceinline__ bf16x16 load_afrag(const __bf16* row, int hi) {
  bf16x8 lo = *(const bf16x8*)(row + hi * 16);
  bf16x8 hh = *(const bf16x8*)(row + hi * 16 + 8);
  return cat16(lo, hh);
}
// B-fragment (32x16; contiguous k order, matching A): lane holds column; 16 contiguous K elems.
__device__ __forceinline__ bf16x16 load_bfrag(const __bf16* p) {
  bf16x8 lo = *(const bf16x8*)(p);
  bf16x8 hh = *(const bf16x8*)(p + 8);
  return cat16(lo, hh);
}

// ----------------- weight transpose + convert: [K][N]f32 -> [L][N][K]bf16 ---
__global__ void wtrans_kernel(const float* __restrict__ W,
                              __bf16* __restrict__ Wt, int K, int N,
                              int total /* L*K*N */) {
  int idx = blockIdx.x * blockDim.x + threadIdx.x;
  if (idx >= total) return;
  int kn = K * N;
  int l = idx / kn;
  int rem = idx - l * kn;
  int n = rem / K;
  int kk = rem - n * K;
  Wt[idx] = (__bf16)W[(size_t)l * kn + (size_t)kk * N + n];
}

// --------------------------- input projection + PE -------------------------
__global__ __launch_bounds__(128) void input_proj_kernel(
    const float* __restrict__ x, const float* __restrict__ Win,
    const float* __restrict__ bin, float* __restrict__ h,
    __bf16* __restrict__ hb) {
  __shared__ float xs[IN_];
  int row = blockIdx.x;
  int t = threadIdx.x;
  if (t < IN_) xs[t] = x[(size_t)row * IN_ + t];
  __syncthreads();
  int s = row % S_;
#pragma unroll
  for (int j = 0; j < D_ / 128; ++j) {
    int n = t + 128 * j;
    float acc = bin[n];
#pragma unroll
    for (int k = 0; k < IN_; ++k) acc += xs[k] * Win[k * D_ + n];
    int i2 = n >> 1;
    float freq = __expf((float)(2 * i2) * -0.017988946f);  // -ln(1e4)/512
    float ang = (float)s * freq;
    acc += (n & 1) ? __cosf(ang) : __sinf(ang);
    size_t o = (size_t)row * D_ + n;
    h[o] = acc;
    hb[o] = (__bf16)acc;
  }
}

// ------------------------------ WMMA GEMM ----------------------------------
// C[M,N] = act(A[M,K](bf16) @ Wt[N,K](bf16)^T + bias[N])
// Block: 128 threads (4 waves). Wave tile 32x64 (2 row-tiles x 4 col-tiles).
// All fragments loaded straight from global as b128 pairs; no LDS.
__global__ __launch_bounds__(128) void wmma_gemm_kernel(
    const __bf16* __restrict__ A, const __bf16* __restrict__ Wt,
    const float* __restrict__ bias, float* __restrict__ outF,
    __bf16* __restrict__ outB, int M, int N, int K, int relu) {
  const int tid  = threadIdx.x;
  const int lane = tid & 31;
  const int wave = tid >> 5;
  const int hi   = lane >> 4;
  const int col  = lane & 15;
  const int m0 = blockIdx.y * 128 + wave * 32;
  const int n0 = blockIdx.x * 64;

  f32x8 acc[2][4];
  f32x8 zero = {0.f, 0.f, 0.f, 0.f, 0.f, 0.f, 0.f, 0.f};
#pragma unroll
  for (int rt = 0; rt < 2; ++rt)
#pragma unroll
    for (int nt = 0; nt < 4; ++nt) acc[rt][nt] = zero;

  const __bf16* arow0 = A + (size_t)(m0 + col) * K;
  const __bf16* arow1 = A + (size_t)(m0 + 16 + col) * K;
  const __bf16* brow0 = Wt + (size_t)(n0 + col) * K + hi * 16;
  const __bf16* brow1 = Wt + (size_t)(n0 + 16 + col) * K + hi * 16;
  const __bf16* brow2 = Wt + (size_t)(n0 + 32 + col) * K + hi * 16;
  const __bf16* brow3 = Wt + (size_t)(n0 + 48 + col) * K + hi * 16;

  for (int kt = 0; kt < K; kt += 32) {
    bf16x16 a0 = load_afrag(arow0 + kt, hi);
    bf16x16 a1 = load_afrag(arow1 + kt, hi);
    bf16x16 b0 = load_bfrag(brow0 + kt);
    bf16x16 b1 = load_bfrag(brow1 + kt);
    bf16x16 b2 = load_bfrag(brow2 + kt);
    bf16x16 b3 = load_bfrag(brow3 + kt);
    acc[0][0] = wmma_bf16(a0, b0, acc[0][0]);
    acc[1][0] = wmma_bf16(a1, b0, acc[1][0]);
    acc[0][1] = wmma_bf16(a0, b1, acc[0][1]);
    acc[1][1] = wmma_bf16(a1, b1, acc[1][1]);
    acc[0][2] = wmma_bf16(a0, b2, acc[0][2]);
    acc[1][2] = wmma_bf16(a1, b2, acc[1][2]);
    acc[0][3] = wmma_bf16(a0, b3, acc[0][3]);
    acc[1][3] = wmma_bf16(a1, b3, acc[1][3]);
  }
#pragma unroll
  for (int rt = 0; rt < 2; ++rt) {
#pragma unroll
    for (int nt = 0; nt < 4; ++nt) {
#pragma unroll
      for (int r = 0; r < 8; ++r) {
        int mm = m0 + rt * 16 + hi * 8 + r;
        int nn = n0 + nt * 16 + col;
        float v = acc[rt][nt][r] + bias[nn];
        if (relu) v = fmaxf(v, 0.f);
        size_t o = (size_t)mm * N + nn;
        if (outF) outF[o] = v;
        if (outB) outB[o] = (__bf16)v;
      }
    }
  }
}

// ----------------- V transpose: [b,s,D] -> vT[b][h][dk][1024] (padded) -----
__global__ void v_transpose_kernel(const __bf16* __restrict__ vb,
                                   __bf16* __restrict__ vT) {
  int idx = blockIdx.x * blockDim.x + threadIdx.x;
  if (idx >= B_ * H_ * DK_ * SROW) return;
  int s  = idx & (SROW - 1);
  int d  = (idx >> 10) & (DK_ - 1);
  int bh = idx >> 16;  // b*H + h
  int b = bh >> 3, h = bh & (H_ - 1);
  vT[idx] = (s < S_) ? vb[(size_t)(b * S_ + s) * D_ + h * DK_ + d]
                     : (__bf16)0.f;
}

// ------------------------------ attention ----------------------------------
// One wave per (b, h, 16-row q tile). Q frags held in regs; K frags direct
// global b128; scores strip in LDS; P frags ds_load_b128; V from vT (b128).
__global__ __launch_bounds__(32) void attention_kernel(
    const __bf16* __restrict__ q, const __bf16* __restrict__ k,
    const __bf16* __restrict__ vT, __bf16* __restrict__ ctx) {
  __shared__ __bf16 sp[16][SROW];  // 32KB: scores -> unnormalized probs
  __shared__ float rowsum[16];
  const int lane = threadIdx.x;
  const int hi = lane >> 4;
  const int col = lane & 15;
  const int q0 = blockIdx.x * 16;
  const int hh = blockIdx.y;
  const int bb = blockIdx.z;
  const size_t headoff = (size_t)bb * S_ * D_ + (size_t)hh * DK_;

  // Q fragments (reused for all key tiles); clamp padded rows
  int qrow = q0 + col;
  if (qrow >= S_) qrow = S_ - 1;
  const __bf16* qp = q + headoff + (size_t)qrow * D_;
  bf16x16 qf0 = load_afrag(qp, hi);        // dims 0..31
  bf16x16 qf1 = load_afrag(qp + 32, hi);   // dims 32..63

  // scores = (Q K^T) / 8
  for (int j = 0; j < STILES; ++j) {
    int key = j * 16 + col;
    int keyc = (key < S_) ? key : (S_ - 1);
    const __bf16* kp = k + headoff + (size_t)keyc * D_ + hi * 16;
    f32x8 acc = {0.f, 0.f, 0.f, 0.f, 0.f, 0.f, 0.f, 0.f};
    acc = wmma_bf16(qf0, load_bfrag(kp), acc);
    acc = wmma_bf16(qf1, load_bfrag(kp + 32), acc);
#pragma unroll
    for (int r = 0; r < 8; ++r) {
      int rw = hi * 8 + r;
      sp[rw][j * 16 + col] = (__bf16)((key < S_) ? acc[r] * 0.125f : -1.0e30f);
    }
  }
  __syncthreads();

  // row softmax (lanes 0..15, one row each); unnormalized probs stay in sp
  if (lane < 16) {
    float mx = -3.0e38f;
    for (int c = 0; c < SPAD; ++c) mx = fmaxf(mx, (float)sp[lane][c]);
    float sum = 0.f;
    for (int c = 0; c < SPAD; ++c) {
      float e = __expf((float)sp[lane][c] - mx);
      sum += e;
      sp[lane][c] = (__bf16)e;
    }
    for (int c = SPAD; c < SROW; ++c) sp[lane][c] = (__bf16)0.f;
    rowsum[lane] = sum;
  }
  __syncthreads();

  // ctx = P @ V : (16 x 1024) x (1024 x 64) via vT[b][h][d][1024]
  const __bf16* vbase = vT + ((size_t)(bb * H_ + hh) * DK_) * SROW;
  f32x8 cacc[4];
  f32x8 zero = {0.f, 0.f, 0.f, 0.f, 0.f, 0.f, 0.f, 0.f};
#pragma unroll
  for (int nt = 0; nt < 4; ++nt) cacc[nt] = zero;

  for (int ks = 0; ks < SROW / 32; ++ks) {
    int base = ks * 32;
    // P fragment from LDS: two aligned 16B chunks
    bf16x8 plo = *(const bf16x8*)&sp[col][base + hi * 16];
    bf16x8 phi = *(const bf16x8*)&sp[col][base + hi * 16 + 8];
    bf16x16 af = cat16(plo, phi);
#pragma unroll
    for (int nt = 0; nt < 4; ++nt) {
      const __bf16* vp = vbase + (size_t)(nt * 16 + col) * SROW + base + hi * 16;
      cacc[nt] = wmma_bf16(af, load_bfrag(vp), cacc[nt]);
    }
  }
#pragma unroll
  for (int nt = 0; nt < 4; ++nt) {
#pragma unroll
    for (int r = 0; r < 8; ++r) {
      int rw = hi * 8 + r;
      int s = q0 + rw;
      if (s < S_) {
        float val = cacc[nt][r] / rowsum[rw];
        ctx[headoff + (size_t)s * D_ + nt * 16 + col] = (__bf16)val;
      }
    }
  }
}

// --------------------- residual add + layernorm ----------------------------
__global__ __launch_bounds__(128) void ln_kernel(
    const float* __restrict__ x, const float* __restrict__ add,
    const float* __restrict__ g, const float* __restrict__ b,
    float* __restrict__ outF, __bf16* __restrict__ outB) {
  __shared__ float red[128];
  int row = blockIdx.x, t = threadIdx.x;
  size_t base = (size_t)row * D_;
  float v[4];
  float s = 0.f;
#pragma unroll
  for (int j = 0; j < 4; ++j) {
    int c = t + 128 * j;
    float val = x[base + c];
    if (add) val += add[base + c];
    v[j] = val;
    s += val;
  }
  red[t] = s;
  __syncthreads();
  for (int st = 64; st > 0; st >>= 1) {
    if (t < st) red[t] += red[t + st];
    __syncthreads();
  }
  float mean = red[0] * (1.f / D_);
  __syncthreads();
  float vs = 0.f;
#pragma unroll
  for (int j = 0; j < 4; ++j) {
    float d = v[j] - mean;
    vs += d * d;
  }
  red[t] = vs;
  __syncthreads();
  for (int st = 64; st > 0; st >>= 1) {
    if (t < st) red[t] += red[t + st];
    __syncthreads();
  }
  float inv = rsqrtf(red[0] * (1.f / D_) + 1e-5f);
#pragma unroll
  for (int j = 0; j < 4; ++j) {
    int c = t + 128 * j;
    float o = (v[j] - mean) * inv * g[c] + b[c];
    outF[base + c] = o;
    if (outB) outB[base + c] = (__bf16)o;
  }
}

// -------------------- naive fp32 dense (heads, ~1% FLOPs) ------------------
__global__ void dense_kernel(const float* __restrict__ A,
                             const float* __restrict__ W,
                             const float* __restrict__ bias,
                             float* __restrict__ out, int M, int N, int K,
                             int act) {
  int idx = blockIdx.x * blockDim.x + threadIdx.x;
  if (idx >= M * N) return;
  int m = idx / N, n = idx - m * N;
  float acc = bias[n];
  const float* a = A + (size_t)m * K;
  for (int kk = 0; kk < K; ++kk) acc += a[kk] * W[(size_t)kk * N + n];
  if (act == 1) acc = fmaxf(acc, 0.f);
  else if (act == 2) acc = 1.f / (1.f + __expf(-acc));
  out[idx] = acc;
}

// ---------------------------------------------------------------------------
extern "C" void kernel_launch(void* const* d_in, const int* in_sizes, int n_in,
                              void* d_out, int out_size, void* d_ws,
                              size_t ws_size, hipStream_t stream) {
  (void)in_sizes; (void)n_in; (void)out_size; (void)ws_size;
  const float* x    = (const float*)d_in[0];
  const float* Win  = (const float*)d_in[1];
  const float* bin  = (const float*)d_in[2];
  const float* Wq   = (const float*)d_in[3];
  const float* bq   = (const float*)d_in[4];
  const float* Wk   = (const float*)d_in[5];
  const float* bk   = (const float*)d_in[6];
  const float* Wv   = (const float*)d_in[7];
  const float* bv   = (const float*)d_in[8];
  const float* Wo   = (const float*)d_in[9];
  const float* bo   = (const float*)d_in[10];
  const float* W1   = (const float*)d_in[11];
  const float* b1   = (const float*)d_in[12];
  const float* W2   = (const float*)d_in[13];
  const float* b2   = (const float*)d_in[14];
  const float* ln1g = (const float*)d_in[15];
  const float* ln1b = (const float*)d_in[16];
  const float* ln2g = (const float*)d_in[17];
  const float* ln2b = (const float*)d_in[18];
  const float* lnfg = (const float*)d_in[19];
  const float* lnfb = (const float*)d_in[20];
  const float* Wf1  = (const float*)d_in[21];
  const float* bf1_ = (const float*)d_in[22];
  const float* Wf2  = (const float*)d_in[23];
  const float* bf2_ = (const float*)d_in[24];
  const float* Wr1  = (const float*)d_in[25];
  const float* br1  = (const float*)d_in[26];
  const float* Wr2  = (const float*)d_in[27];
  const float* br2  = (const float*)d_in[28];
  const float* Wr3  = (const float*)d_in[29];
  const float* br3  = (const float*)d_in[30];
  const float* Wa1  = (const float*)d_in[31];
  const float* ba1  = (const float*)d_in[32];
  const float* Wa2  = (const float*)d_in[33];
  const float* ba2  = (const float*)d_in[34];
  float* out = (float*)d_out;

  char* ws = (char*)d_ws;
  size_t off = 0;
  auto alloc = [&](size_t bytes) -> char* {
    char* p = ws + off;
    off += (bytes + 255) & ~(size_t)255;
    return p;
  };
  // transposed bf16 weights: [L][N][K]
  __bf16* wqT = (__bf16*)alloc((size_t)L_ * D_ * D_ * 2);
  __bf16* wkT = (__bf16*)alloc((size_t)L_ * D_ * D_ * 2);
  __bf16* wvT = (__bf16*)alloc((size_t)L_ * D_ * D_ * 2);
  __bf16* woT = (__bf16*)alloc((size_t)L_ * D_ * D_ * 2);
  __bf16* w1T = (__bf16*)alloc((size_t)L_ * D_ * FF_ * 2);
  __bf16* w2T = (__bf16*)alloc((size_t)L_ * FF_ * D_ * 2);
  float*  hP   = (float*)alloc((size_t)M_ * D_ * 4);
  __bf16* hbP  = (__bf16*)alloc((size_t)M_ * D_ * 2);
  __bf16* qbP  = (__bf16*)alloc((size_t)M_ * D_ * 2);
  __bf16* kbP  = (__bf16*)alloc((size_t)M_ * D_ * 2);
  __bf16* vbP  = (__bf16*)alloc((size_t)M_ * D_ * 2);
  __bf16* vTP  = (__bf16*)alloc((size_t)B_ * H_ * DK_ * SROW * 2);
  __bf16* ctxP = (__bf16*)alloc((size_t)M_ * D_ * 2);
  __bf16* ff1P = (__bf16*)alloc((size_t)M_ * FF_ * 2);
  float*  aF   = (float*)alloc((size_t)M_ * D_ * 4);
  float*  f1P  = (float*)alloc((size_t)M_ * 256 * 4);
  float*  featP= (float*)alloc((size_t)M_ * 128 * 4);
  float*  r1P  = (float*)alloc((size_t)M_ * 256 * 4);
  float*  r2P  = (float*)alloc((size_t)M_ * 512 * 4);
  float*  a1P  = (float*)alloc((size_t)M_ * 64 * 4);

  // 1) weight transpose + bf16 conversion
  {
    int n1 = L_ * D_ * D_;
    int n2 = L_ * D_ * FF_;
    wtrans_kernel<<<(n1 + 255) / 256, 256, 0, stream>>>(Wq, wqT, D_, D_, n1);
    wtrans_kernel<<<(n1 + 255) / 256, 256, 0, stream>>>(Wk, wkT, D_, D_, n1);
    wtrans_kernel<<<(n1 + 255) / 256, 256, 0, stream>>>(Wv, wvT, D_, D_, n1);
    wtrans_kernel<<<(n1 + 255) / 256, 256, 0, stream>>>(Wo, woT, D_, D_, n1);
    wtrans_kernel<<<(n2 + 255) / 256, 256, 0, stream>>>(W1, w1T, D_, FF_, n2);
    wtrans_kernel<<<(n2 + 255) / 256, 256, 0, stream>>>(W2, w2T, FF_, D_, n2);
  }

  // 2) input projection + positional encoding
  input_proj_kernel<<<M_, 128, 0, stream>>>(x, Win, bin, hP, hbP);

  dim3 g512(D_ / 64, M_ / 128);   // (8, 125)
  dim3 gff(FF_ / 64, M_ / 128);   // (32, 125)
  int nvt = B_ * H_ * DK_ * SROW;

  // 3) encoder layers
  for (int l = 0; l < L_; ++l) {
    const __bf16* wq_l = wqT + (size_t)l * D_ * D_;
    const __bf16* wk_l = wkT + (size_t)l * D_ * D_;
    const __bf16* wv_l = wvT + (size_t)l * D_ * D_;
    const __bf16* wo_l = woT + (size_t)l * D_ * D_;
    const __bf16* w1_l = w1T + (size_t)l * D_ * FF_;
    const __bf16* w2_l = w2T + (size_t)l * FF_ * D_;

    wmma_gemm_kernel<<<g512, 128, 0, stream>>>(hbP, wq_l, bq + l * D_,
                                               nullptr, qbP, M_, D_, D_, 0);
    wmma_gemm_kernel<<<g512, 128, 0, stream>>>(hbP, wk_l, bk + l * D_,
                                               nullptr, kbP, M_, D_, D_, 0);
    wmma_gemm_kernel<<<g512, 128, 0, stream>>>(hbP, wv_l, bv + l * D_,
                                               nullptr, vbP, M_, D_, D_, 0);
    v_transpose_kernel<<<(nvt + 255) / 256, 256, 0, stream>>>(vbP, vTP);
    attention_kernel<<<dim3(STILES, H_, B_), 32, 0, stream>>>(qbP, kbP, vTP,
                                                              ctxP);
    wmma_gemm_kernel<<<g512, 128, 0, stream>>>(ctxP, wo_l, bo + l * D_, aF,
                                               nullptr, M_, D_, D_, 0);
    ln_kernel<<<M_, 128, 0, stream>>>(hP, aF, ln1g + l * D_, ln1b + l * D_,
                                      hP, hbP);
    wmma_gemm_kernel<<<gff, 128, 0, stream>>>(hbP, w1_l, b1 + l * FF_,
                                              nullptr, ff1P, M_, FF_, D_, 1);
    wmma_gemm_kernel<<<g512, 128, 0, stream>>>(ff1P, w2_l, b2 + l * D_, aF,
                                               nullptr, M_, D_, FF_, 0);
    ln_kernel<<<M_, 128, 0, stream>>>(hP, aF, ln2g + l * D_, ln2b + l * D_,
                                      hP, hbP);
  }
  // final norm
  ln_kernel<<<M_, 128, 0, stream>>>(hP, nullptr, lnfg, lnfb, hP, nullptr);

  // 4) heads
  auto nd = [&](const float* A, const float* W, const float* bias, float* o,
                int Mx, int Nx, int Kx, int act) {
    int tot = Mx * Nx;
    dense_kernel<<<(tot + 255) / 256, 256, 0, stream>>>(A, W, bias, o, Mx, Nx,
                                                        Kx, act);
  };
  nd(hP, Wf1, bf1_, f1P, M_, 256, 512, 1);
  nd(f1P, Wf2, bf2_, featP, M_, 128, 256, 0);
  nd(featP, Wr1, br1, r1P, M_, 256, 128, 1);
  nd(r1P, Wr2, br2, r2P, M_, 512, 256, 1);
  nd(r2P, Wr3, br3, out, M_, IN_, 512, 0);                 // recon
  nd(featP, Wa1, ba1, a1P, M_, 64, 128, 1);
  nd(a1P, Wa2, ba2, out + (size_t)M_ * IN_, M_, 1, 64, 2); // anomaly
}
